// Net_56599079026987
// MI455X (gfx1250) — matrix operand order repaired and checked by reading, not compiled, not run on hardware
//
#include <hip/hip_runtime.h>
#include <cstddef>

namespace {

constexpr int kN = 20000;   // nodes
constexpr int kE = 640000;  // edges
constexpr int kR = 8;       // relations
constexpr int kH = 128;     // hidden
constexpr int kC = 16;      // classes

typedef __attribute__((ext_vector_type(16))) __bf16 v16bf;
typedef __attribute__((ext_vector_type(8)))  __bf16 v8bf;
typedef __attribute__((ext_vector_type(8)))  float  v8f;

__global__ void zero_f32(float* __restrict__ p, int n) {
  int i = blockIdx.x * blockDim.x + threadIdx.x;
  if (i < n) p[i] = 0.0f;
}

__global__ void count_deg(const int* __restrict__ dst, const int* __restrict__ et,
                          float* __restrict__ cnt, int e) {
  int i = blockIdx.x * blockDim.x + threadIdx.x;
  if (i < e) atomicAdd(&cnt[dst[i] * kR + et[i]], 1.0f);
}

__global__ void make_norm(const int* __restrict__ dst, const int* __restrict__ et,
                          const float* __restrict__ cnt, float* __restrict__ norm, int e) {
  int i = blockIdx.x * blockDim.x + threadIdx.x;
  if (i < e) norm[i] = 1.0f / fmaxf(cnt[dst[i] * kR + et[i]], 1.0f);
}

// Wt[b][o][k] = (b<8 ? Wrel[b][k][o] : Wroot[k][o]) converted to bf16 (transposed
// so a WMMA B-fragment = two contiguous 8-K runs of one output column).
__global__ void convt_w(const float* __restrict__ wrel, const float* __restrict__ wroot,
                        __bf16* __restrict__ wt, int K, int O) {
  int i = blockIdx.x * blockDim.x + threadIdx.x;
  int total = (kR + 1) * K * O;
  if (i >= total) return;
  int b = i / (K * O);
  int rem = i - b * K * O;
  int o = rem / K;
  int k = rem - o * K;
  float w = (b < kR) ? wrel[((size_t)b * K + k) * O + o] : wroot[(size_t)k * O + o];
  wt[i] = (__bf16)w;
}

__global__ void to_bf16(const float* __restrict__ in, __bf16* __restrict__ out, int n) {
  int i = blockIdx.x * blockDim.x + threadIdx.x;
  if (i < n) out[i] = (__bf16)in[i];
}

__global__ void relu_root_bias(float* __restrict__ x, const float* __restrict__ root,
                               const float* __restrict__ bias, int n) {
  int i = blockIdx.x * blockDim.x + threadIdx.x;
  if (i >= n) return;
  float v = x[i] + root[i] + bias[i & (kH - 1)];
  x[i] = v > 0.0f ? v : 0.0f;
}

// One wave per edge: float4 gather of table[r][src][:] (L2-resident), scaled
// scatter-add into out[dst][:]. 4 features per lane.
__global__ void edge_agg_h128(const int* __restrict__ src, const int* __restrict__ dst,
                              const int* __restrict__ et, const float* __restrict__ norm,
                              const float* __restrict__ table, float* __restrict__ out, int e) {
  int wid = (blockIdx.x * blockDim.x + threadIdx.x) >> 5;
  int lane = threadIdx.x & 31;
  if (wid >= e) return;
  int s = src[wid], d = dst[wid], r = et[wid];
  float nn = norm[wid];
  const float4 v = *((const float4*)(table + ((size_t)r * kN + s) * kH) + lane);
  float* op = out + (size_t)d * kH + lane * 4;
  atomicAdd(op + 0, v.x * nn);
  atomicAdd(op + 1, v.y * nn);
  atomicAdd(op + 2, v.z * nn);
  atomicAdd(op + 3, v.w * nn);
}

__global__ void edge_agg_c16(const int* __restrict__ src, const int* __restrict__ dst,
                             const int* __restrict__ et, const float* __restrict__ norm,
                             const float* __restrict__ table, float* __restrict__ out, int e) {
  int gid = blockIdx.x * blockDim.x + threadIdx.x;
  if (gid >= e * kC) return;
  int ed = gid >> 4, c = gid & 15;
  float v = table[((size_t)et[ed] * kN + src[ed]) * kC + c] * norm[ed];
  atomicAdd(&out[dst[ed] * kC + c], v);
}

// Batched bf16 WMMA GEMM, one wave per (batch, M-tile) computing TILES_O 16x16
// output tiles: each A fragment is loaded once and fed to TILES_O WMMAs, cutting
// activation re-reads 8x vs one-tile-per-wave and giving 8 independent WMMA
// chains per K-step.  X:[nodes,128] bf16 row-major, Wt:[batch][O][128] bf16
// (K-contiguous columns), Y:[batch][nodes][O] f32.
template <int TILES_O>
__global__ void wmma_gemm_row(const __bf16* __restrict__ X, const __bf16* __restrict__ Wt,
                              float* __restrict__ Y, int nodes, int batch) {
  constexpr int O = TILES_O * 16;
  const int wid = (blockIdx.x * blockDim.x + threadIdx.x) >> 5;
  const int tilesM = nodes >> 4;
  const int total = batch * tilesM;
  if (wid >= total) return;  // wave-uniform: EXEC stays all-ones for WMMA
  const int lane = threadIdx.x & 31;
  const int hf = lane >> 4;
  const int l16 = lane & 15;
  const int b = wid / tilesM;
  const int tm = wid - b * tilesM;
  const int m0 = tm << 4;

  const __bf16* xrow = X + (size_t)(m0 + l16) * kH;                     // A row m0+l16
  const __bf16* wbase = Wt + ((size_t)b * O + l16) * kH;                // B col l16, tile 0

  v8f acc[TILES_O];
#pragma unroll
  for (int t = 0; t < TILES_O; ++t) acc[t] = {};

#pragma unroll
  for (int kb = 0; kb < kH; kb += 32) {
    // A 16x32 bf16 layout: lane<16 holds K kb..kb+7 then kb+16..kb+23;
    //                      lane>=16 holds K kb+8..kb+15 then kb+24..kb+31.
    const int ka = kb + hf * 8;
    v8bf alo = *(const v8bf*)(xrow + ka);
    v8bf ahi = *(const v8bf*)(xrow + ka + 16);
    v16bf a;
#pragma unroll
    for (int i = 0; i < 8; ++i) { a[i] = alo[i]; a[i + 8] = ahi[i]; }
    // B 32x16 bf16 layout: lane<16 holds K kb..kb+7 then kb+16..kb+23 of its column;
    //                      lane>=16 holds K kb+8..kb+15 then kb+24..kb+31.
    const int kc = kb + hf * 8;
#pragma unroll
    for (int t = 0; t < TILES_O; ++t) {
      const __bf16* wcol = wbase + (size_t)t * 16 * kH + kc;
      v8bf blo = *(const v8bf*)(wcol);
      v8bf bhi = *(const v8bf*)(wcol + 16);
      v16bf bm;
#pragma unroll
      for (int i = 0; i < 8; ++i) { bm[i] = blo[i]; bm[i + 8] = bhi[i]; }
      acc[t] = __builtin_amdgcn_wmma_f32_16x16x32_bf16(false, a, false, bm, (short)0,
                                                       acc[t], false, false);
    }
  }
  // C/D f32 16x16: VGPR v -> row v + hf*8, col l16 (within tile t at col offset t*16)
  float* yrow = Y + ((size_t)b * nodes + m0) * O + l16;
#pragma unroll
  for (int t = 0; t < TILES_O; ++t) {
#pragma unroll
    for (int v = 0; v < 8; ++v) {
      yrow[(size_t)(v + hf * 8) * O + t * 16] = acc[t][v];
    }
  }
}

__global__ void final_lsm(const float* __restrict__ agg, const float* __restrict__ root,
                          const float* __restrict__ bias, float* __restrict__ out, int n) {
  int i = blockIdx.x * blockDim.x + threadIdx.x;
  if (i >= n) return;
  float v[kC];
  float m = -3.4e38f;
#pragma unroll
  for (int c = 0; c < kC; ++c) {
    v[c] = agg[i * kC + c] + root[i * kC + c] + bias[c];
    m = fmaxf(m, v[c]);
  }
  float s = 0.0f;
#pragma unroll
  for (int c = 0; c < kC; ++c) s += expf(v[c] - m);
  float ls = logf(s);
#pragma unroll
  for (int c = 0; c < kC; ++c) out[i * kC + c] = v[c] - m - ls;
}

}  // namespace

extern "C" void kernel_launch(void* const* d_in, const int* in_sizes, int n_in,
                              void* d_out, int out_size, void* d_ws, size_t ws_size,
                              hipStream_t stream) {
  (void)in_sizes; (void)n_in; (void)out_size; (void)ws_size;
  const int*   edge_index = (const int*)d_in[0];
  const int*   etype      = (const int*)d_in[1];
  const float* W0_rel     = (const float*)d_in[2];
  const float* W0_root    = (const float*)d_in[3];
  const float* b0         = (const float*)d_in[4];
  const float* W1_rel     = (const float*)d_in[5];
  const float* W1_root    = (const float*)d_in[6];
  const float* b1         = (const float*)d_in[7];
  const float* W2_rel     = (const float*)d_in[8];
  const float* W2_root    = (const float*)d_in[9];
  const float* b2         = (const float*)d_in[10];
  const int* src = edge_index;
  const int* dst = edge_index + kE;
  float* out = (float*)d_out;

  char* ws = (char*)d_ws;
  size_t off = 0;
  auto alloc = [&](size_t bytes) {
    char* p = ws + off;
    off = (off + bytes + 255) & ~(size_t)255;
    return p;
  };
  float*  cnt  = (float*)alloc((size_t)kN * kR * 4);
  float*  norm = (float*)alloc((size_t)kE * 4);
  float*  bufX = (float*)alloc((size_t)kN * kH * 4);               // agg / activations
  __bf16* xbf  = (__bf16*)alloc((size_t)kN * kH * 2);
  float*  Y    = (float*)alloc((size_t)(kR + 1) * kN * kH * 4);    // xW products (reused L2)
  __bf16* wt1  = (__bf16*)alloc((size_t)(kR + 1) * kH * kH * 2);
  __bf16* wt2  = (__bf16*)alloc((size_t)(kR + 1) * kC * kH * 2);
  float*  agg2 = (float*)alloc((size_t)kN * kC * 4);

  const int B = 256;
  auto cdiv = [](int a, int b) { return (a + b - 1) / b; };

  // edge normalization: 1 / deg(dst, rel)
  zero_f32<<<cdiv(kN * kR, B), B, 0, stream>>>(cnt, kN * kR);
  count_deg<<<cdiv(kE, B), B, 0, stream>>>(dst, etype, cnt, kE);
  make_norm<<<cdiv(kE, B), B, 0, stream>>>(dst, etype, cnt, norm, kE);

  // weight prep: transpose + bf16 (relations 0..7 + root as batch 8)
  convt_w<<<cdiv((kR + 1) * kH * kH, B), B, 0, stream>>>(W1_rel, W1_root, wt1, kH, kH);
  convt_w<<<cdiv((kR + 1) * kH * kC, B), B, 0, stream>>>(W2_rel, W2_root, wt2, kH, kC);

  // ---- layer 0: embedding gather/scatter ----
  zero_f32<<<cdiv(kN * kH, B), B, 0, stream>>>(bufX, kN * kH);
  edge_agg_h128<<<cdiv(kE * 32, B), B, 0, stream>>>(src, dst, etype, norm, W0_rel, bufX, kE);
  relu_root_bias<<<cdiv(kN * kH, B), B, 0, stream>>>(bufX, W0_root, b0, kN * kH);

  // ---- layer 1: batched WMMA GEMM (O=128, 8 tiles/wave) then gather/scatter ----
  to_bf16<<<cdiv(kN * kH, B), B, 0, stream>>>(bufX, xbf, kN * kH);
  {
    int waves = (kR + 1) * (kN / 16);  // 11250 waves, 8 O-tiles each
    wmma_gemm_row<kH / 16><<<cdiv(waves * 32, B), B, 0, stream>>>(xbf, wt1, Y, kN, kR + 1);
  }
  zero_f32<<<cdiv(kN * kH, B), B, 0, stream>>>(bufX, kN * kH);
  edge_agg_h128<<<cdiv(kE * 32, B), B, 0, stream>>>(src, dst, etype, norm, Y, bufX, kE);
  relu_root_bias<<<cdiv(kN * kH, B), B, 0, stream>>>(bufX, Y + (size_t)kR * kN * kH, b1,
                                                     kN * kH);

  // ---- layer 2: batched WMMA GEMM (O=16) then gather/scatter + log_softmax ----
  to_bf16<<<cdiv(kN * kH, B), B, 0, stream>>>(bufX, xbf, kN * kH);
  {
    int waves = (kR + 1) * (kN / 16);  // 11250 waves, 1 O-tile each
    wmma_gemm_row<kC / 16><<<cdiv(waves * 32, B), B, 0, stream>>>(xbf, wt2, Y, kN, kR + 1);
  }
  zero_f32<<<cdiv(kN * kC, B), B, 0, stream>>>(agg2, kN * kC);
  edge_agg_c16<<<cdiv(kE * kC, B), B, 0, stream>>>(src, dst, etype, norm, Y, agg2, kE);
  final_lsm<<<cdiv(kN, B), B, 0, stream>>>(agg2, Y + (size_t)kR * kN * kC, b2, out, kN);
}
